// Decoder_41403484733811
// MI455X (gfx1250) — matrix operand order repaired and checked by reading, not compiled, not run on hardware
//
#include <hip/hip_runtime.h>

#define N_NODES 131072
#define DFEAT   256
#define E_EDGES 262144
#define TOT     (2 * E_EDGES)   // 524288 edges total
#define IN_DIM  512
#define HID     256
#define BROW    520             // padded LDS row stride (shorts): 512 + 8 kills bank conflicts

typedef __attribute__((ext_vector_type(16))) __bf16 bf16x16;
typedef __attribute__((ext_vector_type(8)))  float  f32x8;

union FragU { uint4 u4[2]; bf16x16 v; };

// round-to-nearest-even f32 -> bf16, packed pair into one dword
__device__ __forceinline__ unsigned bf16pk(float lo, float hi) {
    unsigned ul = __float_as_uint(lo);
    unsigned uh = __float_as_uint(hi);
    ul = (ul + 0x7FFFu + ((ul >> 16) & 1u)) >> 16;
    uh = (uh + 0x7FFFu + ((uh >> 16) & 1u)) >> 16;
    return (uh << 16) | (ul & 0xFFFFu);
}

// ---- CDNA5 async global->LDS copy (ASYNCcnt-tracked DMA, bypasses VGPRs) ----
__device__ __forceinline__ void async_g2l_b128(unsigned lds_off, const void* gaddr) {
    asm volatile("global_load_async_to_lds_b128 %0, %1, off"
                 :: "v"(lds_off), "v"(gaddr) : "memory");
}
__device__ __forceinline__ void wait_async0() {
    asm volatile("s_wait_asynccnt 0" ::: "memory");
}

// ---- prep kernel 1: node table f32 -> bf16 (8 elems / thread, 16B stores) ----
__global__ void cvt_nodes_bf16(const float* __restrict__ in,
                               uint4* __restrict__ out, int n8) {
    int t = blockIdx.x * blockDim.x + threadIdx.x;
    if (t >= n8) return;
    const float4* p = (const float4*)in + (size_t)t * 2;
    float4 a = p[0], b = p[1];
    uint4 o;
    o.x = bf16pk(a.x, a.y); o.y = bf16pk(a.z, a.w);
    o.z = bf16pk(b.x, b.y); o.w = bf16pk(b.z, b.w);
    out[t] = o;
}

// ---- prep kernel 2: W1[512][256] f32 -> W1T[256][512] bf16 (transposed) ----
__global__ void cvt_w1t_bf16(const float* __restrict__ w1,
                             unsigned short* __restrict__ w1t) {
    int t = blockIdx.x * blockDim.x + threadIdx.x;   // 0..131071 over [256][512]
    if (t >= HID * IN_DIM) return;
    int n = t >> 9;        // output row (N)
    int k = t & 511;       // output col (K)
    unsigned u = __float_as_uint(w1[k * HID + n]);
    u = (u + 0x7FFFu + ((u >> 16) & 1u)) >> 16;
    w1t[t] = (unsigned short)u;
}

// Stage one W1T N-tile (16 cols x 512 K, 16 KB) into (padded) LDS via async DMA.
// 256 threads x 64 B each; each thread's 32 elements stay within one row.
__device__ __forceinline__ void stage_w1t_tile(const unsigned short* __restrict__ w1t,
                                               unsigned short* sdst, int nt, int tid) {
    const int flat = tid * 32;
    const int nl = flat >> 9;          // 0..15 local column
    const int k  = flat & 511;
    const unsigned short* g = w1t + (((size_t)(nt * 16 + nl)) << 9) + k;
    unsigned short* l = sdst + nl * BROW + k;
    #pragma unroll
    for (int c = 0; c < 4; ++c)
        async_g2l_b128((unsigned)(size_t)(l + c * 8), (const void*)(g + c * 8));
}

// ---- main fused kernel: gather + [128,512]x[512,256] WMMA + relu + dot(W2) ----
template<bool NODES_BF16>
__global__ __launch_bounds__(256, 1) void linkpred_wmma(
    const void* __restrict__ nodes,
    const unsigned short* __restrict__ w1t,   // [HID][IN_DIM] bf16
    const int* __restrict__ pos_src, const int* __restrict__ pos_dst,
    const int* __restrict__ neg_src, const int* __restrict__ neg_dst,
    const float* __restrict__ b1, const float* __restrict__ W2,
    const float* __restrict__ b2,
    float* __restrict__ out)                  // [2*TOT]: logits | labels
{
    __shared__ __align__(16) unsigned short sbuf[2][16 * BROW];  // 2 x 16.25 KB

    const int tid  = threadIdx.x;
    const int lane = tid & 31;
    const int wave = tid >> 5;
    const int tile = blockIdx.x * 8 + wave;   // 16 edges per wave-tile
    const int m    = lane & 15;               // row within tile / local N col
    const int hf   = lane >> 4;               // wave half (A/B layout split)

    // edge row + gathered node indices (pos block first; E%16==0 -> wave-uniform)
    const int  r      = tile * 16 + m;
    const bool isNeg  = (tile * 16) >= E_EDGES;
    const int  e      = isNeg ? (r - E_EDGES) : r;
    const int  srcIdx = isNeg ? neg_src[e] : pos_src[e];
    const int  dstIdx = isNeg ? neg_dst[e] : pos_dst[e];

    // kick off B tile 0 DMA before doing the A gather (overlap)
    stage_w1t_tile(w1t, &sbuf[0][0], 0, tid);

    // Preload all 16 A fragments (K = 512 in steps of 32) in ISA 16-bit A layout:
    //  lanes 0-15 : VGPR0-3 = K[base..base+8), VGPR4-7 = K[base+16..base+24)
    //  lanes 16-31: same shifted by +8
    FragU a[16];
    if constexpr (NODES_BF16) {
        const unsigned short* srcRow = (const unsigned short*)nodes + (size_t)srcIdx * DFEAT;
        const unsigned short* dstRow = (const unsigned short*)nodes + (size_t)dstIdx * DFEAT;
        #pragma unroll
        for (int kk = 0; kk < 16; ++kk) {
            const unsigned short* rowp = (kk < 8) ? srcRow : dstRow;
            const int koff = (kk & 7) * 32 + hf * 8;
            a[kk].u4[0] = *(const uint4*)(rowp + koff);
            a[kk].u4[1] = *(const uint4*)(rowp + koff + 16);
        }
    } else {
        const float* srcRow = (const float*)nodes + (size_t)srcIdx * DFEAT;
        const float* dstRow = (const float*)nodes + (size_t)dstIdx * DFEAT;
        #pragma unroll
        for (int kk = 0; kk < 16; ++kk) {
            const float* rowp = (kk < 8) ? srcRow : dstRow;
            const int koff = (kk & 7) * 32 + hf * 8;
            float4 f0 = *(const float4*)(rowp + koff);
            float4 f1 = *(const float4*)(rowp + koff + 4);
            float4 f2 = *(const float4*)(rowp + koff + 16);
            float4 f3 = *(const float4*)(rowp + koff + 20);
            a[kk].u4[0] = make_uint4(bf16pk(f0.x, f0.y), bf16pk(f0.z, f0.w),
                                     bf16pk(f1.x, f1.y), bf16pk(f1.z, f1.w));
            a[kk].u4[1] = make_uint4(bf16pk(f2.x, f2.y), bf16pk(f2.z, f2.w),
                                     bf16pk(f3.x, f3.y), bf16pk(f3.z, f3.w));
        }
    }

    // Layer-2 partials: acc[j] holds logit contribution of row M = j + 8*hf
    float acc[8];
    #pragma unroll
    for (int j = 0; j < 8; ++j) acc[j] = 0.0f;

    for (int nt = 0; nt < 16; ++nt) {
        wait_async0();       // this wave's DMA slice of buf[nt&1] landed
        __syncthreads();     // everyone's slice landed
        if (nt < 15)         // prefetch next tile into the other buffer
            stage_w1t_tile(w1t, &sbuf[(nt + 1) & 1][0], nt + 1, tid);

        // B fragment: lane holds K of LDS column m in ISA table order
        //  lanes 0-15: K[step..step+8) + K[step+16..step+24), lanes 16-31: same shifted by +8
        const unsigned short* bb = &sbuf[nt & 1][0] + m * BROW + hf * 8;
        f32x8 c = {};
        #pragma unroll
        for (int kk = 0; kk < 16; ++kk) {
            FragU b;
            b.u4[0] = *(const uint4*)(bb + kk * 32);
            b.u4[1] = *(const uint4*)(bb + kk * 32 + 16);
            c = __builtin_amdgcn_wmma_f32_16x16x32_bf16(
                    false, a[kk].v, false, b.v, (short)0, c, false, false);
        }

        // bias + ReLU + fused second layer (H[M,nglob] * W2[nglob])
        const int nglob = nt * 16 + m;
        const float b1v = b1[nglob];
        const float w2v = W2[nglob];
        #pragma unroll
        for (int j = 0; j < 8; ++j) {
            float h = c[j] + b1v;
            h = h > 0.0f ? h : 0.0f;
            acc[j] = fmaf(h, w2v, acc[j]);
        }
        __syncthreads();     // all waves done reading buf[nt&1] before reuse
    }

    // reduce over N (16 lanes of each half; xor<16 never crosses halves)
    #pragma unroll
    for (int j = 0; j < 8; ++j) {
        float v = acc[j];
        v += __shfl_xor(v, 1, 32);
        v += __shfl_xor(v, 2, 32);
        v += __shfl_xor(v, 4, 32);
        v += __shfl_xor(v, 8, 32);
        acc[j] = v;
    }

    const float bias2 = b2[0];
    if (m == 0) {                      // lanes 0 and 16 write rows 0-7 / 8-15
        #pragma unroll
        for (int j = 0; j < 8; ++j) {
            const int idx = tile * 16 + j + hf * 8;
            const float z = acc[j] + bias2;
            out[idx]       = 1.0f / (1.0f + __expf(-z));    // sigmoid logit
            out[TOT + idx] = (idx < E_EDGES) ? 1.0f : 0.0f; // label
        }
    }
}

extern "C" void kernel_launch(void* const* d_in, const int* in_sizes, int n_in,
                              void* d_out, int out_size, void* d_ws, size_t ws_size,
                              hipStream_t stream) {
    const float* block_outputs = (const float*)d_in[0];
    const int*   pos_src = (const int*)d_in[1];
    const int*   pos_dst = (const int*)d_in[2];
    const int*   neg_src = (const int*)d_in[3];
    const int*   neg_dst = (const int*)d_in[4];
    const float* W1 = (const float*)d_in[5];
    const float* b1 = (const float*)d_in[6];
    const float* W2 = (const float*)d_in[7];
    const float* b2 = (const float*)d_in[8];
    float* out = (float*)d_out;

    const size_t nodesBytes = (size_t)N_NODES * DFEAT * sizeof(unsigned short); // 67 MB
    const size_t w1tBytes   = (size_t)HID * IN_DIM * sizeof(unsigned short);    // 256 KB
    const int nBlocks = (TOT / 16) / 8;  // 4096 blocks, 8 wave-tiles (128 edges) each

    if (ws_size >= nodesBytes + w1tBytes) {
        unsigned short* nodesBf = (unsigned short*)d_ws;
        unsigned short* w1t     = (unsigned short*)((char*)d_ws + nodesBytes);
        const int n8 = N_NODES * DFEAT / 8;  // 4,194,304
        cvt_nodes_bf16<<<n8 / 256, 256, 0, stream>>>(block_outputs, (uint4*)nodesBf, n8);
        cvt_w1t_bf16<<<(HID * IN_DIM) / 256, 256, 0, stream>>>(W1, w1t);
        linkpred_wmma<true><<<nBlocks, 256, 0, stream>>>(
            (const void*)nodesBf, w1t, pos_src, pos_dst, neg_src, neg_dst,
            b1, W2, b2, out);
    } else {
        // Fallback: only W1T staged (256 KB); nodes read f32 + packed in-register.
        unsigned short* w1t = (unsigned short*)d_ws;
        cvt_w1t_bf16<<<(HID * IN_DIM) / 256, 256, 0, stream>>>(W1, w1t);
        linkpred_wmma<false><<<nBlocks, 256, 0, stream>>>(
            (const void*)block_outputs, w1t, pos_src, pos_dst, neg_src, neg_dst,
            b1, W2, b2, out);
    }
}
